// FastVPatchCluster_20512763805900
// MI455X (gfx1250) — matrix operand order repaired and checked by reading, not compiled, not run on hardware
//
#include <hip/hip_runtime.h>
#include <hip/hip_bf16.h>
#include <stdint.h>

// DPC-KNN cluster-center selection, N=8192 points, C=128 dims, 256 centers, k=5.
// Tiled recompute of the 8192x8192 distance matrix from the L2-resident bf16
// copy of X via v_wmma_f32_16x16x32_bf16. B tiles (shared by all 4 waves of a
// block) are double-buffered in LDS via global_load_async_to_lds_b128, with
// the copy overlapping the current tile's WMMA work (s_wait_asynccnt fence).
// All O(N^2) reductions operate on raw squared distances (sqrt is monotonic);
// sqrt happens once per row at the end. Row scans are branchless and split
// across two lanes per row (full wave32 utilization), merged via shfl_xor.

#define NPTS   8192
#define CDIM   128
#define NCLUST 256
#define NTILES (NPTS / 16)
#define RSQC   0.08838834764831845f   // 1/sqrt(128)
#define INVC   (1.0f / 128.0f)        // RSQC^2

typedef __attribute__((ext_vector_type(16))) __bf16       v16bf;
typedef __attribute__((ext_vector_type(8)))  float        v8f;
typedef __attribute__((ext_vector_type(4)))  unsigned int v4u;

union BfVec { v16bf v; v4u u[2]; };

// Async copy of 32 bytes (this thread's slice of a 4KB B tile) global -> LDS.
// INST_OFFSET applies to both LDS and global addresses.
__device__ __forceinline__ void async_copy32(unsigned ldsAddr,
                                             const __bf16* gp) {
  unsigned long long ga = (unsigned long long)(uintptr_t)gp;
  asm volatile(
      "global_load_async_to_lds_b128 %0, %1, off\n\t"
      "global_load_async_to_lds_b128 %0, %1, off offset:16"
      :: "v"(ldsAddr), "v"(ga) : "memory");
}

__device__ __forceinline__ void wait_async0() {
  asm volatile("s_wait_asynccnt 0x0" ::: "memory");
}

// Branchless sorted-insert of x into ascending 5-list (bubble network).
__device__ __forceinline__ void ins5(float& t0, float& t1, float& t2,
                                     float& t3, float& t4, float x) {
  float a;
  a = fminf(t0, x); x = fmaxf(t0, x); t0 = a;
  a = fminf(t1, x); x = fmaxf(t1, x); t1 = a;
  a = fminf(t2, x); x = fmaxf(t2, x); t2 = a;
  a = fminf(t3, x); x = fmaxf(t3, x); t3 = a;
  t4 = fminf(t4, x);
}

// ---------------------------------------------------------------------------
// Kernel 1: row squared-norms + bf16 copy of X; init dist2_max slot.
// ---------------------------------------------------------------------------
__global__ __launch_bounds__(256) void prep_kernel(
    const float* __restrict__ X, float* __restrict__ sq,
    __bf16* __restrict__ Xbf, unsigned int* __restrict__ dmaxBits) {
  int r = blockIdx.x * blockDim.x + threadIdx.x;
  if (r == 0) *dmaxBits = 0u;
  if (r < NPTS) {
    const float* xr = X + (size_t)r * CDIM;
    __bf16* br = Xbf + (size_t)r * CDIM;
    float s = 0.0f;
#pragma unroll 4
    for (int c = 0; c < CDIM; ++c) {
      float v = xr[c];
      s += v * v;
      br[c] = (__bf16)v;
    }
    sq[r] = s;
  }
}

// ---------------------------------------------------------------------------
// Tile layouts (per ISA 7.12.2), wave32:
//  A 16x32 bf16: lane m = lane&15, half kh = lane>>4;
//    vec[0..7]  = K(kc*32 + kh*8 + 0..7), vec[8..15] = K(kc*32+16+kh*8+0..7)
//  B 32x16 bf16: lane n = lane&15; vec[0..15] use the same K order as A
//  C v8f: element r at (M = r + 8*kh, N = lane&15)
// Scan split: lane l handles row (l&15), cols kh*8 .. kh*8+7.
// ---------------------------------------------------------------------------

// Kernel 2: density[i] = exp(-mean(d^2 of 5 nearest)) + tie-break noise;
//           global max of raw squared distance via atomicMax on float bits.
__global__ __launch_bounds__(128) void density_kernel(
    const __bf16* __restrict__ Xbf, const float* __restrict__ sq,
    float* __restrict__ den, unsigned int* __restrict__ dmaxBits) {
  __shared__ __align__(16) __bf16 Bstage[2][16 * CDIM];  // 2 x 4KB
  __shared__ float tile[4][16][17];

  const int tid  = threadIdx.x;
  const int wave = tid >> 5;
  const int lane = tid & 31;
  const int lm   = lane & 15;
  const int kh   = lane >> 4;
  const int rowBase = (blockIdx.x * 4 + wave) * 16;
  const int rowA = rowBase + lm;

  BfVec a[4];
#pragma unroll
  for (int kc = 0; kc < 4; ++kc) {
    const __bf16* p = Xbf + (size_t)rowA * CDIM + kc * 32 + kh * 8;
    a[kc].u[0] = *(const v4u*)p;
    a[kc].u[1] = *(const v4u*)(p + 16);
  }
  float sqm[8];
#pragma unroll
  for (int r = 0; r < 8; ++r) sqm[r] = sq[rowBase + kh * 8 + r];

  // Prologue: stage B tile 0 into buffer 0.
  async_copy32((unsigned)(uintptr_t)&Bstage[0][tid * 16], Xbf + tid * 16);
  wait_async0();
  __syncthreads();

  float t0 = 1e30f, t1 = 1e30f, t2 = 1e30f, t3 = 1e30f, t4 = 1e30f;
  float wmax = 0.0f;

  for (int ct = 0; ct < NTILES; ++ct) {
    const int cur = ct & 1;
    // Prefetch next B tile into the other buffer (overlaps WMMA below).
    if (ct + 1 < NTILES) {
      async_copy32((unsigned)(uintptr_t)&Bstage[cur ^ 1][tid * 16],
                   Xbf + (size_t)(ct + 1) * 16 * CDIM + tid * 16);
    }

    // Preload all B fragments, then run the WMMA chain back-to-back.
    BfVec b[4];
#pragma unroll
    for (int kc = 0; kc < 4; ++kc) {
      const __bf16* bp = &Bstage[cur][lm * CDIM + kc * 32 + kh * 8];
      b[kc].u[0] = *(const v4u*)bp;        // ds_load_b128
      b[kc].u[1] = *(const v4u*)(bp + 16); // ds_load_b128
    }
    v8f c = {0.f, 0.f, 0.f, 0.f, 0.f, 0.f, 0.f, 0.f};
#pragma unroll
    for (int kc = 0; kc < 4; ++kc) {
      c = __builtin_amdgcn_wmma_f32_16x16x32_bf16(
          false, a[kc].v, false, b[kc].v, (short)0, c, false, false);
    }

    const int colBase = ct * 16;
    const float sqn = sq[colBase + lm];
#pragma unroll
    for (int r = 0; r < 8; ++r) {
      float d2 = fmaxf(sqm[r] + sqn - 2.0f * c[r], 0.0f);  // raw squared dist
      tile[wave][kh * 8 + r][lm] = d2;
      wmax = fmaxf(wmax, d2);
    }

    wait_async0();     // next B tile resident (this wave's slice)
    __syncthreads();   // all slices resident; dist2 tile visible; readers done

    // Branchless scan, split across two lanes per row.
#pragma unroll
    for (int n = 0; n < 8; ++n) {
      ins5(t0, t1, t2, t3, t4, tile[wave][lm][kh * 8 + n]);
    }
  }

  // Merge the two half-row 5-lists (partner lane = lane ^ 16).
#pragma unroll
  for (int j = 0; j < 5; ++j) {
    float p0 = __shfl_xor(t0, 16, 32);
    float p1 = __shfl_xor(t1, 16, 32);
    float p2 = __shfl_xor(t2, 16, 32);
    float p3 = __shfl_xor(t3, 16, 32);
    float p4 = __shfl_xor(t4, 16, 32);
    (void)p1; (void)p2; (void)p3; (void)p4;
    // insert one partner element per pass, rotating (p0 always smallest left)
    ins5(t0, t1, t2, t3, t4, j == 0 ? p0 : (j == 1 ? p1 : (j == 2 ? p2 : (j == 3 ? p3 : p4))));
  }

#pragma unroll
  for (int off = 16; off > 0; off >>= 1)
    wmax = fmaxf(wmax, __shfl_xor(wmax, off, 32));
  if (lane == 0) atomicMax(dmaxBits, __float_as_uint(wmax));

  if (lane < 16) {
    int row = rowBase + lane;
    // mean of squared scaled distances of the 5 nearest = 0.2*sum(d2)/C
    float mean = (t0 + t1 + t2 + t3 + t4) * 0.2f * INVC;
    unsigned h = (unsigned)row * 2654435761u;
    h ^= h >> 16; h *= 2246822519u; h ^= h >> 13;
    float noise = (float)(h & 0xFFFFFFu) * (5.9604645e-8f * 1e-6f);
    den[row] = __expf(-mean) + noise;
  }
}

// Kernel 3: dist_parent[i] = min over {j : den[j] > den[i]} dist(i,j), else
//           dist_max (all in raw d^2 space); score[i]=sqrt(min d2)*RSQC*den[i].
__global__ __launch_bounds__(128) void parent_kernel(
    const __bf16* __restrict__ Xbf, const float* __restrict__ sq,
    const float* __restrict__ den, const unsigned int* __restrict__ dmaxBits,
    float* __restrict__ score) {
  __shared__ __align__(16) __bf16 Bstage[2][16 * CDIM];
  __shared__ float tile[4][16][17];
  __shared__ float dtile[4][16];

  const int tid  = threadIdx.x;
  const int wave = tid >> 5;
  const int lane = tid & 31;
  const int lm   = lane & 15;
  const int kh   = lane >> 4;
  const int rowBase = (blockIdx.x * 4 + wave) * 16;
  const int rowA = rowBase + lm;

  BfVec a[4];
#pragma unroll
  for (int kc = 0; kc < 4; ++kc) {
    const __bf16* p = Xbf + (size_t)rowA * CDIM + kc * 32 + kh * 8;
    a[kc].u[0] = *(const v4u*)p;
    a[kc].u[1] = *(const v4u*)(p + 16);
  }
  float sqm[8];
#pragma unroll
  for (int r = 0; r < 8; ++r) sqm[r] = sq[rowBase + kh * 8 + r];

  const float dmax2 = __uint_as_float(*dmaxBits);  // max raw squared dist
  float minv = dmax2;
  const float myden = den[rowBase + lm];           // both half-lanes need it

  async_copy32((unsigned)(uintptr_t)&Bstage[0][tid * 16], Xbf + tid * 16);
  wait_async0();
  __syncthreads();

  for (int ct = 0; ct < NTILES; ++ct) {
    const int cur = ct & 1;
    if (ct + 1 < NTILES) {
      async_copy32((unsigned)(uintptr_t)&Bstage[cur ^ 1][tid * 16],
                   Xbf + (size_t)(ct + 1) * 16 * CDIM + tid * 16);
    }

    BfVec b[4];
#pragma unroll
    for (int kc = 0; kc < 4; ++kc) {
      const __bf16* bp = &Bstage[cur][lm * CDIM + kc * 32 + kh * 8];
      b[kc].u[0] = *(const v4u*)bp;
      b[kc].u[1] = *(const v4u*)(bp + 16);
    }
    v8f c = {0.f, 0.f, 0.f, 0.f, 0.f, 0.f, 0.f, 0.f};
#pragma unroll
    for (int kc = 0; kc < 4; ++kc) {
      c = __builtin_amdgcn_wmma_f32_16x16x32_bf16(
          false, a[kc].v, false, b[kc].v, (short)0, c, false, false);
    }

    const int colBase = ct * 16;
    const float sqn = sq[colBase + lm];
    dtile[wave][lm] = den[colBase + lm];
#pragma unroll
    for (int r = 0; r < 8; ++r) {
      tile[wave][kh * 8 + r][lm] = fmaxf(sqm[r] + sqn - 2.0f * c[r], 0.0f);
    }

    wait_async0();
    __syncthreads();

    // Masked min, split across two lanes per row; select instead of branch.
#pragma unroll
    for (int n = 0; n < 8; ++n) {
      int col = kh * 8 + n;
      float cand = (dtile[wave][col] > myden) ? tile[wave][lm][col] : 1e30f;
      minv = fminf(minv, cand);
    }
  }

  minv = fminf(minv, __shfl_xor(minv, 16, 32));  // merge half-rows

  if (lane < 16) {
    float dist_parent = sqrtf(minv) * RSQC;  // one sqrt per row
    score[rowBase + lane] = dist_parent * myden;
  }
}

// Kernel 4: top-256 by score, descending, lower index wins ties (matches
// jax.lax.top_k stability). Single block, scores staged in LDS.
__global__ __launch_bounds__(1024) void topk_kernel(
    const float* __restrict__ score, int* __restrict__ out) {
  __shared__ float s[NPTS];
  __shared__ float rv[1024];
  __shared__ int   ri[1024];

  for (int i = threadIdx.x; i < NPTS; i += 1024) s[i] = score[i];
  __syncthreads();

  for (int sel = 0; sel < NCLUST; ++sel) {
    float best = -1e30f;
    int bi = NPTS;
    for (int i = threadIdx.x; i < NPTS; i += 1024) {
      float v = s[i];
      if (v > best || (v == best && i < bi)) { best = v; bi = i; }
    }
    rv[threadIdx.x] = best;
    ri[threadIdx.x] = bi;
    __syncthreads();
    for (int off = 512; off > 0; off >>= 1) {
      if (threadIdx.x < off) {
        float ov = rv[threadIdx.x + off];
        int   oi = ri[threadIdx.x + off];
        if (ov > rv[threadIdx.x] ||
            (ov == rv[threadIdx.x] && oi < ri[threadIdx.x])) {
          rv[threadIdx.x] = ov;
          ri[threadIdx.x] = oi;
        }
      }
      __syncthreads();
    }
    if (threadIdx.x == 0) {
      out[sel] = ri[0];
      s[ri[0]] = -1e30f;
    }
    __syncthreads();
  }
}

// ---------------------------------------------------------------------------
extern "C" void kernel_launch(void* const* d_in, const int* in_sizes, int n_in,
                              void* d_out, int out_size, void* d_ws, size_t ws_size,
                              hipStream_t stream) {
  const float* X = (const float*)d_in[0];   // [1, 8192, 128] f32

  // Workspace layout (floats): sq[N] | den[N] | score[N] | dmaxBits | Xbf
  float* w = (float*)d_ws;
  float*        sq       = w;
  float*        den      = w + NPTS;
  float*        score    = w + 2 * NPTS;
  unsigned int* dmaxBits = (unsigned int*)(w + 3 * NPTS);
  __bf16*       Xbf      = (__bf16*)(w + 4 * NPTS);   // 16B-aligned; 2MB

  int* out = (int*)d_out;   // 256 int32 indices

  prep_kernel<<<NPTS / 256, 256, 0, stream>>>(X, sq, Xbf, dmaxBits);
  density_kernel<<<NPTS / 64, 128, 0, stream>>>(Xbf, sq, den, dmaxBits);
  parent_kernel<<<NPTS / 64, 128, 0, stream>>>(Xbf, sq, den, dmaxBits, score);
  topk_kernel<<<1, 1024, 0, stream>>>(score, out);
}
